// RNN2D_64957085385347
// MI455X (gfx1250) — compile-verified, matrix operand order checked by reading, not run on hardware
//
#include <hip/hip_runtime.h>
#include <hip/hip_bf16.h>

typedef __attribute__((ext_vector_type(16))) __bf16 v16bf;
typedef __attribute__((ext_vector_type(8)))  float  v8f;

#define UDIM 512
#define LDIM 16
#define BW   16      // batch rows per workgroup (WMMA M)
#define NTPW 4       // 16-wide N tiles per wave
#define NWAVES 8

// LDS byte offsets (dynamic shared)
#define OFF_STATES 0                         // 16 cols * 16 ktiles * 32 lanes * 16 bf16 = 262144 B
#define OFF_H      262144                    // 2 * 8192 bf16 = 32768 B
#define OFF_SPIN   294912                    // 16 batch * 16 rows * u32 (bit col) = 1024 B
#define OFF_LOG    295936                    // 32 f32 = 128 B
#define SMEM_BYTES 296064                    // < 320KB WGP cap

__device__ __forceinline__ unsigned short f2bf(float f) {
    unsigned u = __float_as_uint(f);
    u += 0x7FFFu + ((u >> 16) & 1u);         // round-to-nearest-even
    return (unsigned short)(u >> 16);
}

// Re-tile a (K=512, N=512) f32 weight into WMMA-B bf16 lane layout:
// tile t = kt*32+nt holds K=kt*32..+31, N=nt*16..+15; lane l slot e:
//   N = nt*16 + (l&15),  K = kt*32 + (e/8)*16 + (l/16)*8 + (e&7)
__global__ void __launch_bounds__(256) rnn2d_prep_kernel(
    const float* __restrict__ Wh, const float* __restrict__ Wv,
    unsigned short* __restrict__ outH, unsigned short* __restrict__ outV) {
    int idx  = blockIdx.x * 256 + threadIdx.x;          // 0..262143
    int e    = idx & 15;
    int lane = (idx >> 4) & 31;
    int t    = idx >> 9;                                // 0..511
    int kt   = t >> 5, nt = t & 31;
    int n = nt * 16 + (lane & 15);
    int k = kt * 32 + ((e >> 3) << 4) + ((lane >> 4) << 3) + (e & 7);
    outH[idx] = f2bf(Wh[k * UDIM + n]);
    outV[idx] = f2bf(Wv[k * UDIM + n]);
}

__global__ void __launch_bounds__(256, 1) rnn2d_main_kernel(
    const int*   __restrict__ x,
    const float* __restrict__ Wih,  const float* __restrict__ Wiv,
    const float* __restrict__ bch,
    const float* __restrict__ Wout, const float* __restrict__ bout,
    const unsigned short* __restrict__ WhT,
    const unsigned short* __restrict__ WvT,
    float* __restrict__ out) {
    extern __shared__ char smem[];
    unsigned short* sStates = (unsigned short*)(smem + OFF_STATES);
    unsigned short* sH      = (unsigned short*)(smem + OFF_H);
    unsigned*       sSpin   = (unsigned*)      (smem + OFF_SPIN);   // [b][row] bit=col
    float*          sLog    = (float*)         (smem + OFF_LOG);

    const int tid  = threadIdx.x;
    const int lane = tid & 31;
    const int wave = tid >> 5;
    const int wg   = blockIdx.x;
    const int hi   = lane >> 4;     // lane-group: C/D rows m = r + 8*hi
    const int ln   = lane & 15;     // C/D column within tile

    // ---- hoist all per-n small weights into registers (D=2 => one-hot GEMMs
    //      degenerate to a 2-way select; W_out head is 2 columns) ----
    float w0h[NTPW], w1h[NTPW], w0v[NTPW], w1v[NTPW], bn[NTPW], wo0[NTPW], wo1[NTPW];
#pragma unroll
    for (int nt = 0; nt < NTPW; ++nt) {
        const int n = (wave * NTPW + nt) * 16 + ln;
        w0h[nt] = Wih[n];  w1h[nt] = Wih[UDIM + n];
        w0v[nt] = Wiv[n];  w1v[nt] = Wiv[UDIM + n];
        bn[nt]  = bch[n];
        wo0[nt] = Wout[n * 2 + 0];  wo1[nt] = Wout[n * 2 + 1];
    }
    const float b_out0 = bout[0], b_out1 = bout[1];

    // ---- one-time LDS init ----
    {   // bit-pack the spin lattice: one u32 per (batch,row), bit c = x[b,row,c]
        const int b = tid >> 4, row = tid & 15;
        const int* xr = x + (wg * BW + b) * (LDIM * LDIM) + row * LDIM;
        unsigned w = 0;
#pragma unroll
        for (int cc = 0; cc < LDIM; ++cc) w |= ((unsigned)(xr[cc] & 1)) << cc;
        sSpin[tid] = w;
    }
    for (int i = tid; i < 16 * 8192; i += 256) sStates[i] = 0;   // zero row-above states
    for (int i = tid; i < 2 * 8192;  i += 256) sH[i] = 0;
    if (tid < 32) sLog[tid] = 0.f;
    __syncthreads();

    float logp_tot = 0.f;           // live in wave0 lanes 0..15 (m = lane)
    int par = 0;

    for (int row = 0; row < LDIM; ++row) {
        const int dir = (row & 1) ? -1 : 1;
        // reset h at row start (read buffer of cell 0)
        for (int i = tid; i < 8192; i += 256) sH[par * 8192 + i] = 0;

        // per-row spin words for this lane's 8 batch rows:
        // bits [15:0] = current row, bits [31:16] = row above
        unsigned spw[8];
#pragma unroll
        for (int r = 0; r < 8; ++r) {
            const int m = r + 8 * hi;
            const unsigned cur  = sSpin[m * 16 + row];
            const unsigned prev = (row > 0) ? sSpin[m * 16 + row - 1] : 0u;
            spw[r] = (cur & 0xFFFFu) | (prev << 16);
        }
        __syncthreads();

        for (int c = 0; c < LDIM; ++c) {
            const int col  = (dir > 0) ? c : (LDIM - 1 - c);
            const int colL = col - dir;   // previous cell in scan order

            // ---- C init: bias + one-hot row selects (pure VALU) ----
            v8f acc[NTPW];
#pragma unroll
            for (int nt = 0; nt < NTPW; ++nt) {
#pragma unroll
                for (int r = 0; r < 8; ++r) {
                    float v = bn[nt];
                    if (c > 0)   v += ((spw[r] >> colL) & 1u)        ? w1h[nt] : w0h[nt];
                    if (row > 0) v += ((spw[r] >> (16 + col)) & 1u)  ? w1v[nt] : w0v[nt];
                    acc[nt][r] = v;
                }
            }

            // ---- recurrent GEMMs: acc += h@W_carry_h + cv@W_carry_v (bf16 WMMA) ----
            const unsigned short* hbase  = sH + par * 8192;
            const unsigned short* cvbase = sStates + col * 8192;
#pragma unroll 4
            for (int kt = 0; kt < 16; ++kt) {
                v16bf aH = *(const v16bf*)(hbase  + (kt * 32 + lane) * 16);
                v16bf aV = *(const v16bf*)(cvbase + (kt * 32 + lane) * 16);
                if (kt < 15) {  // pull next K-slab of weights toward L0/L2
                    __builtin_prefetch(WhT + ((kt + 1) * 32 + wave * NTPW) * 512, 0, 1);
                    __builtin_prefetch(WvT + ((kt + 1) * 32 + wave * NTPW) * 512, 0, 1);
                }
#pragma unroll
                for (int nt = 0; nt < NTPW; ++nt) {
                    const int t = kt * 32 + (wave * NTPW + nt);
                    v16bf bH = *(const v16bf*)(WhT + (t * 32 + lane) * 16);
                    v16bf bV = *(const v16bf*)(WvT + (t * 32 + lane) * 16);
                    acc[nt] = __builtin_amdgcn_wmma_f32_16x16x32_bf16(
                        false, aH, false, bH, (short)0, acc[nt], false, false);
                    acc[nt] = __builtin_amdgcn_wmma_f32_16x16x32_bf16(
                        false, aV, false, bV, (short)0, acc[nt], false, false);
                }
            }
            __syncthreads();   // all waves done READING h[par] & states[col]

            // ---- ELU, scatter new hidden to LDS (A layout), output-head partials ----
            unsigned short* hdst = sH + (par ^ 1) * 8192;
            unsigned short* cdst = sStates + col * 8192;
            float p0[8], p1[8];
#pragma unroll
            for (int r = 0; r < 8; ++r) { p0[r] = 0.f; p1[r] = 0.f; }
#pragma unroll
            for (int nt = 0; nt < NTPW; ++nt) {
                const int n = (wave * NTPW + nt) * 16 + ln;
#pragma unroll
                for (int r = 0; r < 8; ++r) {
                    float v = acc[nt][r];
                    v = (v > 0.f) ? v : (__expf(v) - 1.f);          // ELU
                    p0[r] += v * wo0[nt]; p1[r] += v * wo1[nt];
                    const int m = r + 8 * hi;
                    const int kl = n & 31, kt2 = n >> 5;
                    const int hi2 = (kl >> 3) & 1;
                    const int e2  = ((kl >> 4) << 3) + (kl & 7);
                    const int off = (kt2 * 32 + (m + 16 * hi2)) * 16 + e2;
                    const unsigned short bv = f2bf(v);
                    hdst[off] = bv;          // next cell's h
                    cdst[off] = bv;          // next row's cv at this column
                }
            }
            // reduce logits over the 16 N-columns held by this lane group
#pragma unroll
            for (int s = 1; s < 16; s <<= 1) {
#pragma unroll
                for (int r = 0; r < 8; ++r) {
                    p0[r] += __shfl_xor(p0[r], s, 32);
                    p1[r] += __shfl_xor(p1[r], s, 32);
                }
            }
            if (ln == 0) {
#pragma unroll
                for (int r = 0; r < 8; ++r) {
                    const int m = r + 8 * hi;
                    atomicAdd(&sLog[m * 2 + 0], p0[r]);
                    atomicAdd(&sLog[m * 2 + 1], p1[r]);
                }
            }
            __syncthreads();   // logits complete; h/state stores visible

            if (wave == 0 && lane < 16) {
                const int m = lane;
                const float l0 = sLog[m * 2 + 0] + b_out0;
                const float l1 = sLog[m * 2 + 1] + b_out1;
                const int  sc = (sSpin[m * 16 + row] >> col) & 1u;
                const float mx  = fmaxf(l0, l1);
                const float lse = mx + __logf(__expf(l0 - mx) + __expf(l1 - mx));
                logp_tot += (sc ? l1 : l0) - lse;
                sLog[m * 2 + 0] = 0.f; sLog[m * 2 + 1] = 0.f;
            }
            __syncthreads();   // logits reset before next cell's atomics
            par ^= 1;
        }
    }

    if (wave == 0 && lane < 16) {
        float v = logp_tot;
        if (!(v == v)) v = 0.f;               // nan_to_num
        out[wg * BW + lane] = v;
    }
}

extern "C" void kernel_launch(void* const* d_in, const int* in_sizes, int n_in,
                              void* d_out, int out_size, void* d_ws, size_t ws_size,
                              hipStream_t stream) {
    (void)in_sizes; (void)n_in; (void)out_size; (void)ws_size;
    const int*   x    = (const int*)  d_in[0];
    const float* Wih  = (const float*)d_in[1];
    const float* Wiv  = (const float*)d_in[2];
    const float* Wch  = (const float*)d_in[3];
    const float* bch  = (const float*)d_in[4];
    const float* Wcv  = (const float*)d_in[5];
    const float* Wout = (const float*)d_in[6];
    const float* bout = (const float*)d_in[7];

    unsigned short* WhT = (unsigned short*)d_ws;           // 512KB
    unsigned short* WvT = WhT + 512 * 512;                 // 512KB

    rnn2d_prep_kernel<<<1024, 256, 0, stream>>>(Wch, Wcv, WhT, WvT);
    rnn2d_main_kernel<<<16, 256, SMEM_BYTES, stream>>>(
        x, Wih, Wiv, bch, Wout, bout, WhT, WvT, (float*)d_out);
}
